// WorldModelRNN_42021960024500
// MI455X (gfx1250) — matrix operand order repaired and checked by reading, not compiled, not run on hardware
//
#include <hip/hip_runtime.h>
#include <math.h>

// ---------------- problem constants ----------------
#define HIST   10
#define LAY    8
#define H      256
#define BATCH  512
#define RSTEPS 64          // num_rollout - 1
#define TT     76          // HIST + R + 1
#define STATE  12
#define ACT    4
#define OUTD   9
#define MLP_H  15000
#define K1     160         // 13*HIST=130 padded to mult of 32
#define KMLP   15008       // MLP_H padded to mult of 32
#define NH2    4096        // L*2*H
#define GATES  1024        // 4*H
#define HPAD   264         // padded LDS row stride for h

typedef __attribute__((ext_vector_type(16))) __bf16 v16bf;
typedef __attribute__((ext_vector_type(8)))  float  v8f;

__device__ __forceinline__ __bf16 f2bf(float f) {
  unsigned u; __builtin_memcpy(&u, &f, 4);
  unsigned r = u + 0x7FFFu + ((u >> 16) & 1u);   // round-to-nearest-even
  unsigned short hs = (unsigned short)(r >> 16);
  __bf16 b; __builtin_memcpy(&b, &hs, 2);
  return b;
}

__device__ __forceinline__ v8f zero8() { v8f z = {}; return z; }

__device__ __forceinline__ float sigm(float x) { return 1.f / (1.f + __expf(-x)); }

// A fragment (16x32 bf16, M rows x K): lanes 0-15 rows M=lane, K {0..15};
// lanes 16-31 same rows, K {16..31}.  X points at the M-tile origin.
__device__ __forceinline__ v16bf load_a16(const __bf16* X, int ld, int mlane,
                                          int khalf /*0 or 8*/, int kbase) {
  const __bf16* p = X + (size_t)mlane * ld + kbase + khalf;
  v16bf a;
#pragma unroll
  for (int e = 0; e < 16; e++) { a[e] = p[khalf + e]; }
  return a;
}

// B fragment (32x16 bf16, K x N) from row-major W[n][k]: lane holds column
// n = ntile*16 + (lane&15); lanes 0-15 carry K kbase..+15, lanes 16-31 +16..+31
// => one contiguous 32B run per lane.
__device__ __forceinline__ v16bf load_b16(const __bf16* W, int ldk, int nrow,
                                          int ksel /* kbase + 16*(lane>=16) */) {
  const __bf16* p = W + (size_t)nrow * ldk + ksel;
  v16bf b;
#pragma unroll
  for (int e = 0; e < 16; e++) b[e] = p[e];
  return b;
}

__device__ __forceinline__ v8f wmma_bf16(v16bf a, v16bf b, v8f c) {
  return __builtin_amdgcn_wmma_f32_16x16x32_bf16(false, a, false, b, (short)0, c,
                                                 false, false);
}

// ---------------- prep kernels ----------------
__global__ void k_build_enc(__bf16* enc, const float* states, const float* acts) {
  int i = blockIdx.x * blockDim.x + threadIdx.x;
  if (i >= BATCH * K1) return;
  int b = i / K1, c = i % K1;
  float v = 0.f;
  if (c < 13 * HIST) {
    int t = c / 13, j = c % 13;
    v = (j < 9) ? states[((size_t)b * TT + t) * STATE + 3 + j]
                : acts[((size_t)b * TT + t) * ACT + (j - 9)];
  }
  enc[i] = f2bf(v);
}

// fp32 [rs x cs] -> bf16 [rd x cd], zero padded
__global__ void k_pad_cvt(__bf16* dst, const float* src, int rd, int cd, int rs, int cs) {
  long total = (long)rd * cd;
  for (long i = blockIdx.x * (long)blockDim.x + threadIdx.x; i < total;
       i += (long)gridDim.x * blockDim.x) {
    int r = (int)(i / cd), c = (int)(i % cd);
    float v = (r < rs && c < cs) ? src[(long)r * cs + c] : 0.f;
    dst[i] = f2bf(v);
  }
}

__global__ void k_bias(float* dst, float* bfc_pad, const float* bi0, const float* bh0,
                       const float* bi, const float* bh, const float* bfc) {
  int i = blockIdx.x * blockDim.x + threadIdx.x;
  if (i < LAY * GATES) {
    int l = i >> 10, n = i & 1023;
    dst[i] = (l == 0) ? (bi0[n] + bh0[n])
                      : (bi[(l - 1) * GATES + n] + bh[(l - 1) * GATES + n]);
  }
  if (i < 16) bfc_pad[i] = (i < OUTD) ? bfc[i] : 0.f;
}

// ---------------- encoder GEMMs ----------------
// mlp[512][15008] = relu(enc[512][160] @ W1^T + b1), bf16 out
__global__ __launch_bounds__(128) void k_gemm1(const __bf16* enc, const __bf16* w1,
                                               const float* b1, __bf16* mlp) {
  int lane = threadIdx.x & 31, wv = threadIdx.x >> 5;
  int mt = blockIdx.x;
  int nt = blockIdx.y * 4 + wv;
  if (nt >= KMLP / 16) return;                 // 938 N tiles
  int mlane = lane & 15, hi = lane >> 4;
  v8f acc = zero8();
  const __bf16* A = enc + (size_t)mt * 16 * K1;
  for (int kt = 0; kt < K1 / 32; kt++) {
    v16bf a = load_a16(A, K1, mlane, hi * 8, kt * 32);
    v16bf b = load_b16(w1, K1, nt * 16 + mlane, kt * 32 + hi * 16);
    acc = wmma_bf16(a, b, acc);
  }
  int col = nt * 16 + mlane;
  float bias = (col < MLP_H) ? b1[col] : 0.f;
#pragma unroll
  for (int r = 0; r < 8; r++) {
    float v = acc[r] + bias;
    v = v > 0.f ? v : 0.f;
    if (col >= MLP_H) v = 0.f;
    mlp[(size_t)(mt * 16 + hi * 8 + r) * KMLP + col] = f2bf(v);
  }
}

// hc[512][4096] = mlp @ W2^T + b2 (fp32 out).  4 M-tiles x 2 N-tiles per wave.
__global__ __launch_bounds__(128) void k_gemm2(const __bf16* mlp, const __bf16* w2,
                                               const float* b2, float* hc) {
  int lane = threadIdx.x & 31, wv = threadIdx.x >> 5;
  int mlane = lane & 15, hi = lane >> 4;
  int mt0 = blockIdx.x * 4;                    // 32 M tiles / 8 blocks
  int nt0 = blockIdx.y * 8 + wv * 2;           // 256 N tiles / 32 blocks
  v8f acc[4][2];
#pragma unroll
  for (int i = 0; i < 4; i++)
#pragma unroll
    for (int j = 0; j < 2; j++) acc[i][j] = zero8();

  for (int kt = 0; kt < KMLP / 32; kt++) {
    v16bf bf0 = load_b16(w2, KMLP, (nt0 + 0) * 16 + mlane, kt * 32 + hi * 16);
    v16bf bf1 = load_b16(w2, KMLP, (nt0 + 1) * 16 + mlane, kt * 32 + hi * 16);
#pragma unroll
    for (int i = 0; i < 4; i++) {
      v16bf a = load_a16(mlp + (size_t)(mt0 + i) * 16 * KMLP, KMLP, mlane, hi * 8, kt * 32);
      acc[i][0] = wmma_bf16(a, bf0, acc[i][0]);
      acc[i][1] = wmma_bf16(a, bf1, acc[i][1]);
    }
  }
#pragma unroll
  for (int i = 0; i < 4; i++)
#pragma unroll
    for (int j = 0; j < 2; j++) {
      int col = (nt0 + j) * 16 + mlane;
      float bias = b2[col];
#pragma unroll
      for (int r = 0; r < 8; r++)
        hc[(size_t)((mt0 + i) * 16 + hi * 8 + r) * NH2 + col] = acc[i][j][r] + bias;
    }
}

// ---------------- persistent rollout kernel ----------------
// 32 blocks x 512 threads (16 waves).  Block owns 16 batch rows; h[8] lives in
// LDS (bf16), c[8] lives in per-wave registers.  All 64 steps in one launch.
__global__ __launch_bounds__(512) void k_rollout(
    const float* hc, const float* states, const float* acts, const float* dts,
    const __bf16* wi0, const __bf16* wih, const __bf16* whh, const float* bias,
    const __bf16* wfc, const float* bfcp, float* out) {
  __shared__ __bf16 h_lds[LAY][16][HPAD];
  __shared__ __bf16 x0[16][32];
  __shared__ float out_s[16][16];

  const int tid = threadIdx.x;
  const int lane = tid & 31, wv = tid >> 5;     // 16 waves
  const int mlane = lane & 15, hi = lane >> 4;
  const int b0 = blockIdx.x * 16;
  const int ccol = wv * 16 + mlane;             // this wave's h column (0..255)

  // init h (LDS, bf16) from encoder output hc[b][l*2H + c]
  for (int i = tid; i < LAY * 16 * H; i += blockDim.x) {
    int l = i / (16 * H), rem = i % (16 * H);
    int m = rem / H, c = rem % H;
    h_lds[l][m][c] = f2bf(hc[(size_t)(b0 + m) * NH2 + l * 2 * H + c]);
  }
  for (int i = tid; i < 16 * 32; i += blockDim.x) x0[i / 32][i % 32] = f2bf(0.f);

  // init c (registers, fp32) in D-fragment layout
  v8f creg[LAY];
#pragma unroll
  for (int l = 0; l < LAY; l++)
#pragma unroll
    for (int r = 0; r < 8; r++) {
      int m = hi * 8 + r;
      creg[l][r] = hc[(size_t)(b0 + m) * NH2 + l * 2 * H + H + ccol];
    }
  __syncthreads();

  for (int s = 0; s < RSTEPS; s++) {
    // build layer-0 input x = [dt, pred(9), act(4)], zero-padded to 32
    for (int i = tid; i < 16 * 14; i += blockDim.x) {
      int m = i / 14, c = i % 14, b = b0 + m;
      float v;
      if (c == 0)       v = dts[(size_t)b * TT + (HIST + 1 + s)];
      else if (c < 10)  v = (s == 0) ? states[((size_t)b * TT + HIST) * STATE + 3 + (c - 1)]
                                     : out_s[m][c - 1];
      else              v = acts[((size_t)b * TT + HIST + s) * ACT + (c - 10)];
      x0[m][c] = f2bf(v);
    }
    __syncthreads();

#pragma unroll
    for (int l = 0; l < LAY; l++) {
      const __bf16* Wi  = (l == 0) ? wi0 : (wih + (size_t)(l - 1) * GATES * H);
      const __bf16* Wh  = whh + (size_t)l * GATES * H;
      const __bf16* X   = (l == 0) ? &x0[0][0] : &h_lds[l - 1][0][0];
      const int ldx     = (l == 0) ? 32 : HPAD;
      const int ldwi    = (l == 0) ? 32 : H;
      const int kxsteps = (l == 0) ? 1 : 8;

      // gates i/f/g/o at this wave's columns: N tiles g*16 + wv (of 64)
      v8f acc[4];
#pragma unroll
      for (int g = 0; g < 4; g++) {
        float bv = bias[l * GATES + g * H + ccol];
#pragma unroll
        for (int r = 0; r < 8; r++) acc[g][r] = bv;
      }
      for (int kt = 0; kt < kxsteps; kt++) {
        v16bf a = load_a16(X, ldx, mlane, hi * 8, kt * 32);
#pragma unroll
        for (int g = 0; g < 4; g++) {
          v16bf b = load_b16(Wi, ldwi, g * H + wv * 16 + mlane, kt * 32 + hi * 16);
          acc[g] = wmma_bf16(a, b, acc[g]);
        }
      }
      for (int kt = 0; kt < 8; kt++) {
        v16bf a = load_a16(&h_lds[l][0][0], HPAD, mlane, hi * 8, kt * 32);
#pragma unroll
        for (int g = 0; g < 4; g++) {
          v16bf b = load_b16(Wh, H, g * H + wv * 16 + mlane, kt * 32 + hi * 16);
          acc[g] = wmma_bf16(a, b, acc[g]);
        }
      }
      __syncthreads();   // all waves done READING h_lds[l] / x before overwrite

#pragma unroll
      for (int r = 0; r < 8; r++) {
        float ig = sigm(acc[0][r]);
        float fg = sigm(acc[1][r]);
        float gg = tanhf(acc[2][r]);
        float og = sigm(acc[3][r]);
        float cn = fg * creg[l][r] + ig * gg;
        creg[l][r] = cn;
        h_lds[l][hi * 8 + r][ccol] = f2bf(og * tanhf(cn));
      }
      __syncthreads();   // h_lds[l] visible for next layer's x
    }

    // fc head: one 16x16 tile, wave 0 only (wave-uniform branch, EXEC all-1s)
    if (wv == 0) {
      v8f acc;
      float bv = bfcp[mlane];
#pragma unroll
      for (int r = 0; r < 8; r++) acc[r] = bv;
      for (int kt = 0; kt < 8; kt++) {
        v16bf a = load_a16(&h_lds[LAY - 1][0][0], HPAD, mlane, hi * 8, kt * 32);
        v16bf b = load_b16(wfc, H, mlane, kt * 32 + hi * 16);
        acc = wmma_bf16(a, b, acc);
      }
#pragma unroll
      for (int r = 0; r < 8; r++) {
        int m = hi * 8 + r;
        out_s[m][mlane] = acc[r];
        if (mlane < OUTD)
          out[((size_t)(b0 + m) * RSTEPS + s) * OUTD + mlane] = acc[r];
      }
    }
    __syncthreads();
  }
}

// ---------------- host launcher ----------------
extern "C" void kernel_launch(void* const* d_in, const int* in_sizes, int n_in,
                              void* d_out, int out_size, void* d_ws, size_t ws_size,
                              hipStream_t stream) {
  const float* states = (const float*)d_in[0];
  const float* acts   = (const float*)d_in[1];
  const float* dts    = (const float*)d_in[2];
  const float* W1     = (const float*)d_in[3];
  const float* b1     = (const float*)d_in[4];
  const float* W2     = (const float*)d_in[5];
  const float* b2     = (const float*)d_in[6];
  const float* W_ih0  = (const float*)d_in[7];
  const float* W_hh0  = (const float*)d_in[8];
  const float* b_ih0  = (const float*)d_in[9];
  const float* b_hh0  = (const float*)d_in[10];
  const float* W_ih   = (const float*)d_in[11];
  const float* W_hh   = (const float*)d_in[12];
  const float* b_ih   = (const float*)d_in[13];
  const float* b_hh   = (const float*)d_in[14];
  const float* Wfc    = (const float*)d_in[15];
  const float* bfc    = (const float*)d_in[16];
  (void)in_sizes; (void)n_in; (void)out_size; (void)ws_size;

  char* p = (char*)d_ws;
  size_t off = 0;
  auto take = [&](size_t bytes) -> char* {
    char* r = p + off;
    off += (bytes + 255) & ~(size_t)255;
    return r;
  };

  __bf16* enc_bf = (__bf16*)take((size_t)BATCH * K1 * 2);
  __bf16* w1bf   = (__bf16*)take((size_t)KMLP /*15008 rows*/ * 160 * 2);
  __bf16* mlp    = (__bf16*)take((size_t)BATCH * KMLP * 2);
  __bf16* w2bf   = (__bf16*)take((size_t)NH2 * KMLP * 2);
  float*  hc     = (float*)take((size_t)BATCH * NH2 * 4);
  __bf16* wi0b   = (__bf16*)take((size_t)GATES * 32 * 2);
  __bf16* wihb   = (__bf16*)take((size_t)7 * GATES * H * 2);
  __bf16* whhb   = (__bf16*)take((size_t)LAY * GATES * H * 2);
  float*  biasb  = (float*)take((size_t)LAY * GATES * 4);
  __bf16* wfcb   = (__bf16*)take((size_t)16 * H * 2);
  float*  bfcp   = (float*)take(16 * 4);

  // encoder input + weight conversion
  k_build_enc<<<(BATCH * K1 + 255) / 256, 256, 0, stream>>>(enc_bf, states, acts);
  k_pad_cvt<<<2048, 256, 0, stream>>>(w1bf, W1, KMLP /*15008 padded rows*/, K1, MLP_H, 130);
  k_gemm1<<<dim3(BATCH / 16, 235), 128, 0, stream>>>(enc_bf, w1bf, b1, mlp);
  k_pad_cvt<<<8192, 256, 0, stream>>>(w2bf, W2, NH2, KMLP, NH2, MLP_H);
  k_gemm2<<<dim3(8, 32), 128, 0, stream>>>(mlp, w2bf, b2, hc);

  // LSTM weight packing (bf16, row-major [n][k])
  k_pad_cvt<<<256, 256, 0, stream>>>(wi0b, W_ih0, GATES, 32, GATES, 14);
  k_pad_cvt<<<2048, 256, 0, stream>>>(wihb, W_ih, 7 * GATES, H, 7 * GATES, H);
  k_pad_cvt<<<1024, 256, 0, stream>>>(whhb, W_hh0, GATES, H, GATES, H);
  k_pad_cvt<<<2048, 256, 0, stream>>>(whhb + (size_t)GATES * H, W_hh, 7 * GATES, H,
                                      7 * GATES, H);
  k_pad_cvt<<<32, 256, 0, stream>>>(wfcb, Wfc, 16, H, OUTD, H);
  k_bias<<<(LAY * GATES + 255) / 256, 256, 0, stream>>>(biasb, bfcp, b_ih0, b_hh0,
                                                        b_ih, b_hh, bfc);

  // single persistent kernel for all 64 rollout steps
  k_rollout<<<BATCH / 16, 512, 0, stream>>>(hc, states, acts, dts, wi0b, wihb, whhb,
                                            biasb, wfcb, bfcp, (float*)d_out);
}
